// ConditionalNormalizingFlow_13932873909154
// MI455X (gfx1250) — compile-verified, matrix operand order checked by reading, not run on hardware
//
#include <hip/hip_runtime.h>
#include <hip/hip_bf16.h>

// ---------------------------------------------------------------------------
// ConditionalNormalizingFlow for MI455X (gfx1250, wave32, WMMA, async-to-LDS
// in both directions: global->LDS tile staging and LDS->global tile drain)
// B=32768, D=64, C=64, H=1024, L=6, SPLIT=32, NIN=96
// ---------------------------------------------------------------------------

#define NF_B     32768
#define NF_D     64
#define NF_C     64
#define NF_H     1024
#define NF_L     6
#define NF_NIN   96
#define NF_SPLIT 32
#define NF_EPS   1e-5f

typedef __attribute__((ext_vector_type(16))) __bf16 v16bf;
typedef __attribute__((ext_vector_type(8)))  float  v8f;

// On gfx1250 flat addressing, a generic pointer to LDS carries the LDS byte
// offset in its low 32 bits.
__device__ __forceinline__ unsigned nf_lds_addr(const void* p) {
    return (unsigned)(size_t)p;
}

// 16-byte global -> LDS async copy (ASYNCcnt-tracked, no VGPR data movement).
__device__ __forceinline__ void nf_async_cp16(unsigned lds, const void* g) {
    asm volatile("global_load_async_to_lds_b128 %0, %1, off"
                 :: "v"(lds), "v"((unsigned long long)(size_t)g)
                 : "memory");
}

// 16-byte LDS -> global async store (ASYNCcnt-tracked).
__device__ __forceinline__ void nf_async_st16(void* g, unsigned lds) {
    asm volatile("global_store_async_from_lds_b128 %0, %1, off"
                 :: "v"((unsigned long long)(size_t)g), "v"(lds)
                 : "memory");
}

__device__ __forceinline__ void nf_wait_async() {
    asm volatile("s_wait_asynccnt 0x0" ::: "memory");
}

// ---------------------------------------------------------------------------
// Weight prepack: f32 [K][N] (optionally split over two K x nsplit sources)
// -> bf16 [N][K] in workspace. Runs once per launch; makes every GEMM tile a
// contiguous 16B async copy and halves weight read traffic.
// ---------------------------------------------------------------------------
__global__ __launch_bounds__(256)
void nf_prepack(const float* __restrict__ Wa, const float* __restrict__ Wb,
                __bf16* __restrict__ out, int K, int N, int nsplit,
                int ldwa, int ldwb)
{
    const int idx = blockIdx.x * 256 + threadIdx.x;   // n*K + k
    if (idx >= K * N) return;
    const int n = idx / K, k = idx - n * K;
    const float v = (n < nsplit) ? Wa[(size_t)k * ldwa + n]
                                 : Wb[(size_t)k * ldwb + (n - nsplit)];
    out[idx] = (__bf16)v;
}

// ---------------------------------------------------------------------------
// WMMA GEMM:  out(M x N) = act( A(M x K) @ W(K x N) + bias )
//   - A: bf16 buffer (h1/h2) staged via async-to-LDS, or assembled on the fly
//     from [x[:,keep], cond] (f32 -> bf16 gather) when A_FROM_X.
//   - W: prepacked bf16 [N][K], staged via async-to-LDS.
//   - Double-buffered LDS input tiles: stage k+1 while WMMAs run on k.
//   - Output tile staged in LDS (aliasing the stage buffers) and drained with
//     coalesced global_store_async_from_lds_b128.
//   - SPLITB: bias from two 32-wide vectors (bs | bt).
// BM=128, BK=32 fixed; 256 threads = 8 wave32 waves.
// ---------------------------------------------------------------------------
template <int BN, int WAVES_M, int WAVES_N, bool A_FROM_X, bool SPLITB, bool BF16_RELU_OUT>
__global__ __launch_bounds__(256)
void nf_gemm_wmma(const __bf16* __restrict__ Ah,   // bf16 activations (if !A_FROM_X)
                  const float* __restrict__ xbuf,  // x state (if A_FROM_X)
                  const float* __restrict__ cond,  // cond  (if A_FROM_X)
                  int keep_par,
                  const __bf16* __restrict__ Wp,   // prepacked bf16 [N][K]
                  const float* __restrict__ ba, const float* __restrict__ bb,
                  __bf16* __restrict__ outb, float* __restrict__ outf,
                  int K, int lda, int ldc)
{
    constexpr int BM = 128, BK = 32, PK = 40;        // PK: padded LDS K-stride
    constexpr int WMr = BM / WAVES_M;
    constexpr int WNr = BN / WAVES_N;
    constexpr int MT = WMr / 16, NT = WNr / 16;

    // shared pool: double-buffered input tiles, aliased by the output tile
    constexpr int AST = BM * PK;                     // A stage elems
    constexpr int BST = BN * PK;                     // B stage elems
    constexpr int STAGE_BYTES = (2 * AST + 2 * BST) * 2;
    constexpr int OELEM = BF16_RELU_OUT ? 2 : 4;     // out element bytes
    constexpr int ORS = BN + (BF16_RELU_OUT ? 8 : 4);// out row stride (elems)
    constexpr int OUT_BYTES = BM * ORS * OELEM;      // = 34816 both configs
    constexpr int SMEM_BYTES = STAGE_BYTES > OUT_BYTES ? STAGE_BYTES : OUT_BYTES;
    __shared__ __align__(16) char smem[SMEM_BYTES];
    // NOTE: do not build constant pointer tables from LDS addresses (the
    // addrspacecast ends up in a static initializer ld.lld rejects) —
    // derive buffer pointers on the fly instead.
    auto AsBuf = [&](int i) -> __bf16* { return (__bf16*)smem + i * AST; };
    auto BsBuf = [&](int i) -> __bf16* { return (__bf16*)smem + 2 * AST + i * BST; };

    const int tid  = threadIdx.x;
    const int wave = tid >> 5;
    const int lane = tid & 31;
    const int wm   = wave % WAVES_M;
    const int wn   = wave / WAVES_M;
    const int rowBase = blockIdx.x * BM;
    const int colBase = blockIdx.y * BN;
    const int lhalf = lane >> 4;      // K half: 0 -> K 0..15, 1 -> K 16..31
    const int lm    = lane & 15;      // row (A) / col (B) within 16

    v8f acc[MT][NT];
    for (int mt = 0; mt < MT; ++mt)
        for (int nt = 0; nt < NT; ++nt)
            for (int e = 0; e < 8; ++e) acc[mt][nt][e] = 0.0f;

    // stage one BK-slice of A into LDS buffer
    auto stageA = [&](int k0, __bf16* dst) {
        if constexpr (A_FROM_X) {
            // gather keep-columns / cond, convert f32 -> bf16 (VALU path)
            const int r  = tid >> 1;
            const int c0 = (tid & 1) * 16;
            const int grow = rowBase + r;
            for (int c = 0; c < 16; ++c) {
                const int kg = k0 + c0 + c;
                float v;
                if (kg < NF_SPLIT) v = xbuf[(size_t)grow * NF_D + keep_par + 2 * kg];
                else               v = cond[(size_t)grow * NF_C + (kg - NF_SPLIT)];
                dst[r * PK + c0 + c] = (__bf16)v;
            }
        } else {
            // 128 rows x 64B -> 512 async 16B chunks, 2 per thread
            for (int p = 0; p < 2; ++p) {
                const int idx = p * 256 + tid;
                const int r = idx >> 2, ch = idx & 3;
                nf_async_cp16(nf_lds_addr(dst + r * PK + ch * 8),
                              Ah + (size_t)(rowBase + r) * lda + k0 + ch * 8);
            }
        }
    };
    // stage one BK-slice of W ([N][K] prepacked) into LDS buffer
    auto stageB = [&](int k0, __bf16* dst) {
        constexpr int CHUNKS = BN * 4;               // BN rows x 4x16B
        for (int idx = tid; idx < CHUNKS; idx += 256) {
            const int r = idx >> 2, ch = idx & 3;
            nf_async_cp16(nf_lds_addr(dst + r * PK + ch * 8),
                          Wp + (size_t)(colBase + r) * K + k0 + ch * 8);
        }
    };

    // prologue: stage slice 0
    stageA(0, AsBuf(0));
    stageB(0, BsBuf(0));
    nf_wait_async();
    __syncthreads();

    const int nk = K / BK;
    for (int t = 0; t < nk; ++t) {
        const __bf16* curA = AsBuf(t & 1);
        const __bf16* curB = BsBuf(t & 1);
        const bool more = (t + 1 < nk);
        if (more) {                                  // prefetch next slice
            stageA((t + 1) * BK, AsBuf((t + 1) & 1));
            stageB((t + 1) * BK, BsBuf((t + 1) & 1));
        }

        // fragments (2 x float4 -> v16bf, 16B-aligned LDS reads)
        union Frag { v16bf v; float4 f[2]; };
        Frag a[MT], b[NT];
        for (int mt = 0; mt < MT; ++mt) {
            const __bf16* p = &curA[(wm * WMr + mt * 16 + lm) * PK + lhalf * 16];
            a[mt].f[0] = *(const float4*)(p);
            a[mt].f[1] = *(const float4*)(p + 8);
        }
        for (int nt = 0; nt < NT; ++nt) {
            const __bf16* p = &curB[(wn * WNr + nt * 16 + lm) * PK + lhalf * 16];
            b[nt].f[0] = *(const float4*)(p);
            b[nt].f[1] = *(const float4*)(p + 8);
        }

        // 16x16x32 bf16 WMMA, f32 accumulate
        for (int mt = 0; mt < MT; ++mt)
            for (int nt = 0; nt < NT; ++nt)
                acc[mt][nt] = __builtin_amdgcn_wmma_f32_16x16x32_bf16(
                    false, a[mt].v, false, b[nt].v, (short)0, acc[mt][nt],
                    false, false);

        if (more) {
            nf_wait_async();      // next slice resident in LDS
            __syncthreads();      // all waves done reading cur / writing nxt
        }
    }

    // ---- epilogue: bias (+relu), stage tile in LDS, async drain to global ----
    // C layout: VGPR r, lanes 0-15 -> M=r, N=lane; lanes 16-31 -> M=r+8
    __syncthreads();              // input-tile reads done; reuse smem as out tile
    for (int nt = 0; nt < NT; ++nt) {
        const int nloc = wn * WNr + nt * 16 + lm;
        float bias;
        if constexpr (SPLITB) bias = (nloc < NF_SPLIT) ? ba[nloc] : bb[nloc - NF_SPLIT];
        else                  bias = ba[colBase + nloc];
        for (int mt = 0; mt < MT; ++mt) {
            const int mloc = wm * WMr + mt * 16 + 8 * lhalf;
            for (int r = 0; r < 8; ++r) {
                float v = acc[mt][nt][r] + bias;
                if constexpr (BF16_RELU_OUT) {
                    v = v > 0.0f ? v : 0.0f;
                    ((__bf16*)smem)[(mloc + r) * ORS + nloc] = (__bf16)v;
                } else {
                    ((float*)smem)[(mloc + r) * ORS + nloc] = v;
                }
            }
        }
    }
    __syncthreads();
    {
        constexpr int CPR = (BN * OELEM) / 16;       // 16B chunks per row (=16)
        char* const gout = BF16_RELU_OUT ? (char*)outb : (char*)outf;
        for (int idx = tid; idx < BM * CPR; idx += 256) {
            const int r = idx / CPR, ch = idx % CPR;
            char* gdst = gout + ((size_t)(rowBase + r) * ldc + colBase) * OELEM + ch * 16;
            nf_async_st16(gdst, nf_lds_addr((const char*)smem + r * ORS * OELEM + ch * 16));
        }
        nf_wait_async();          // drain before wave exit
    }
}

// ---------------------------------------------------------------------------
// init: x = z, logdet = 0, stats block = 0
// stats layout (floats): [0..63]=sum, [64..127]=sumsq, [128..191]=scale,
//                        [192..255]=shift, [256]=ld_const
// ---------------------------------------------------------------------------
__global__ __launch_bounds__(256)
void nf_init(const float* __restrict__ z, float* __restrict__ xbuf,
             float* __restrict__ logdet, float* __restrict__ stats)
{
    const int idx = blockIdx.x * 256 + threadIdx.x;
    if (idx < NF_B * NF_D) xbuf[idx] = z[idx];
    if (idx < NF_B)        logdet[idx] = 0.0f;
    if (idx < 512)         stats[idx] = 0.0f;
}

// ---------------------------------------------------------------------------
// coupling: x_chg = x_chg * exp(tanh(s_raw)) + t ; logdet += sum(log_s)
// ---------------------------------------------------------------------------
__global__ __launch_bounds__(256)
void nf_couple(const float* __restrict__ st, float* __restrict__ xbuf,
               float* __restrict__ logdet, int chg_par)
{
    const int r = blockIdx.x * 256 + threadIdx.x;
    float ld = 0.0f;
    const size_t rb = (size_t)r * NF_D;
    for (int j = 0; j < NF_SPLIT; ++j) {
        const float ls = tanhf(st[rb + j]);
        const float tv = st[rb + NF_SPLIT + j];
        const int c = chg_par + 2 * j;
        const float xv = xbuf[rb + c];
        xbuf[rb + c] = xv * __expf(ls) + tv;
        ld += ls;
    }
    logdet[r] += ld;
}

// ---------------------------------------------------------------------------
// batch stats: per-column sum / sumsq over B rows (coalesced, LDS reduce,
// global f32 atomics into stats[0..127]). 256 rows per block.
// ---------------------------------------------------------------------------
__global__ __launch_bounds__(256)
void nf_stats(const float* __restrict__ xbuf, float* __restrict__ stats)
{
    const int tid = threadIdx.x;
    const int col = tid & 63, quad = tid >> 6;
    const int base = blockIdx.x * 256;
    float s = 0.0f, ss = 0.0f;
    for (int it = 0; it < 64; ++it) {
        const float v = xbuf[(size_t)(base + it * 4 + quad) * NF_D + col];
        s += v; ss += v * v;
    }
    __shared__ float red[4][64];
    red[quad][col] = s;
    __syncthreads();
    if (quad == 0)
        atomicAdd(&stats[col], red[0][col] + red[1][col] + red[2][col] + red[3][col]);
    __syncthreads();
    red[quad][col] = ss;
    __syncthreads();
    if (quad == 0)
        atomicAdd(&stats[64 + col], red[0][col] + red[1][col] + red[2][col] + red[3][col]);
}

// ---------------------------------------------------------------------------
// finalize BN: scale/shift per column, ld_const += sum(log|bn_w|),
// re-zero accumulators for the next layer (keeps each launch deterministic).
// ---------------------------------------------------------------------------
__global__ void nf_finalize(float* __restrict__ stats,
                            const float* __restrict__ bn_w,
                            const float* __restrict__ bn_b)
{
    const int c = threadIdx.x;   // 64 threads
    const float inv_b = 1.0f / (float)NF_B;
    const float mean = stats[c] * inv_b;
    const float var  = stats[64 + c] * inv_b - mean * mean;
    const float w = bn_w[c];
    const float sc = w * rsqrtf(var + NF_EPS);
    stats[128 + c] = sc;
    stats[192 + c] = bn_b[c] - mean * sc;
    atomicAdd(&stats[256], logf(fabsf(w)));
    stats[c] = 0.0f;
    stats[64 + c] = 0.0f;
}

// ---------------------------------------------------------------------------
// BN apply in place: x = x * scale[col] + shift[col]   (float4 vectorized)
// ---------------------------------------------------------------------------
__global__ __launch_bounds__(256)
void nf_bn_apply(float* __restrict__ xbuf, const float* __restrict__ stats)
{
    const int idx = blockIdx.x * 256 + threadIdx.x;   // over B*16 float4
    float4 v = ((const float4*)xbuf)[idx];
    const int c = (idx & 15) * 4;
    v.x = v.x * stats[128 + c + 0] + stats[192 + c + 0];
    v.y = v.y * stats[128 + c + 1] + stats[192 + c + 1];
    v.z = v.z * stats[128 + c + 2] + stats[192 + c + 2];
    v.w = v.w * stats[128 + c + 3] + stats[192 + c + 3];
    ((float4*)xbuf)[idx] = v;
}

// ---------------------------------------------------------------------------
// pack outputs: out[0 : B*D) = x ; out[B*D + r] = logdet[r] + ld_const
// ---------------------------------------------------------------------------
__global__ __launch_bounds__(256)
void nf_pack(const float* __restrict__ xbuf, const float* __restrict__ logdet,
             const float* __restrict__ stats, float* __restrict__ out)
{
    const int idx = blockIdx.x * 256 + threadIdx.x;
    if (idx < NF_B * NF_D)            out[idx] = xbuf[idx];
    else if (idx < NF_B * (NF_D + 1)) out[idx] = logdet[idx - NF_B * NF_D] + stats[256];
}

// ---------------------------------------------------------------------------
extern "C" void kernel_launch(void* const* d_in, const int* in_sizes, int n_in,
                              void* d_out, int out_size, void* d_ws, size_t ws_size,
                              hipStream_t stream)
{
    const float* z    = (const float*)d_in[0];
    const float* cond = (const float*)d_in[1];
    const float* W1   = (const float*)d_in[2];
    const float* b1   = (const float*)d_in[3];
    const float* W2   = (const float*)d_in[4];
    const float* b2   = (const float*)d_in[5];
    const float* Ws   = (const float*)d_in[6];
    const float* bs   = (const float*)d_in[7];
    const float* Wt   = (const float*)d_in[8];
    const float* bt   = (const float*)d_in[9];
    const float* bnw  = (const float*)d_in[10];
    const float* bnb  = (const float*)d_in[11];
    float* out = (float*)d_out;

    // workspace layout
    char* ws = (char*)d_ws;
    float*  xbuf   = (float*)(ws);                              // 8 MB
    float*  logdet = (float*)(ws + (size_t)8  * 1024 * 1024);   // 128 KB
    float*  stats  = (float*)(ws + (size_t)8  * 1024 * 1024 + 128 * 1024);
    __bf16* h1     = (__bf16*)(ws + (size_t)16 * 1024 * 1024);  // 64 MB
    __bf16* h2     = (__bf16*)(ws + (size_t)80 * 1024 * 1024);  // 64 MB
    float*  st     = (float*)h1;   // overlay: h1 dead when st is produced
    __bf16* wpack  = (__bf16*)(ws + (size_t)144 * 1024 * 1024); // ~14.6 MB

    const size_t W1T_E  = (size_t)NF_H * NF_NIN;          //  98304
    const size_t W2T_E  = (size_t)NF_H * NF_H;            // 1048576
    const size_t WSTT_E = (size_t)(2 * NF_SPLIT) * NF_H;  //   65536
    const size_t PER_L  = W1T_E + W2T_E + WSTT_E;

    // ---- prepack all weights: f32 [K][N] -> bf16 [N][K] ----
    for (int i = 0; i < NF_L; ++i) {
        __bf16* w1t = wpack + (size_t)i * PER_L;
        __bf16* w2t = w1t + W1T_E;
        __bf16* wst = w2t + W2T_E;
        nf_prepack<<<(int)(W1T_E / 256), 256, 0, stream>>>(
            W1 + (size_t)i * NF_NIN * NF_H, nullptr, w1t,
            NF_NIN, NF_H, NF_H, NF_H, 0);
        nf_prepack<<<(int)(W2T_E / 256), 256, 0, stream>>>(
            W2 + (size_t)i * NF_H * NF_H, nullptr, w2t,
            NF_H, NF_H, NF_H, NF_H, 0);
        nf_prepack<<<(int)(WSTT_E / 256), 256, 0, stream>>>(
            Ws + (size_t)i * NF_H * NF_SPLIT, Wt + (size_t)i * NF_H * NF_SPLIT,
            wst, NF_H, 2 * NF_SPLIT, NF_SPLIT, NF_SPLIT, NF_SPLIT);
    }

    nf_init<<<(NF_B * NF_D + 255) / 256, 256, 0, stream>>>(z, xbuf, logdet, stats);

    for (int i = 0; i < NF_L; ++i) {
        const int keep_par = (i % 2 == 0) ? 0 : 1;
        const int chg_par  = 1 - keep_par;
        const __bf16* w1t = wpack + (size_t)i * PER_L;
        const __bf16* w2t = w1t + W1T_E;
        const __bf16* wst = w2t + W2T_E;

        // h1 = relu([x_keep, cond] @ W1 + b1)        (B x 96) @ (96 x 1024)
        nf_gemm_wmma<128, 2, 4, true, false, true>
            <<<dim3(NF_B / 128, NF_H / 128), 256, 0, stream>>>(
                nullptr, xbuf, cond, keep_par, w1t,
                b1 + (size_t)i * NF_H, nullptr,
                h1, nullptr, NF_NIN, 0, NF_H);

        // h2 = relu(h1 @ W2 + b2)                    (B x 1024) @ (1024 x 1024)
        nf_gemm_wmma<128, 2, 4, false, false, true>
            <<<dim3(NF_B / 128, NF_H / 128), 256, 0, stream>>>(
                h1, nullptr, nullptr, 0, w2t,
                b2 + (size_t)i * NF_H, nullptr,
                h2, nullptr, NF_H, NF_H, NF_H);

        // st = h2 @ [Ws | Wt] + [bs | bt]            (B x 1024) @ (1024 x 64)
        nf_gemm_wmma<64, 4, 2, false, true, false>
            <<<dim3(NF_B / 128, 1), 256, 0, stream>>>(
                h2, nullptr, nullptr, 0, wst,
                bs + (size_t)i * NF_SPLIT, bt + (size_t)i * NF_SPLIT,
                nullptr, st, NF_H, NF_H, 2 * NF_SPLIT);

        // coupling update + per-row log_det
        nf_couple<<<NF_B / 256, 256, 0, stream>>>(st, xbuf, logdet, chg_par);

        // batchnorm (training): stats -> scale/shift -> apply
        nf_stats<<<NF_B / 256, 256, 0, stream>>>(xbuf, stats);
        nf_finalize<<<1, 64, 0, stream>>>(stats, bnw + i * NF_D, bnb + i * NF_D);
        nf_bn_apply<<<(NF_B * 16) / 256, 256, 0, stream>>>(xbuf, stats);
    }

    nf_pack<<<(NF_B * (NF_D + 1) + 255) / 256, 256, 0, stream>>>(xbuf, logdet, stats, out);
}
